// DeeperGCN_37941741093106
// MI455X (gfx1250) — matrix operand order repaired and checked by reading, not compiled, not run on hardware
//
#include <hip/hip_runtime.h>
#include <math.h>

#define N_NODES 50000
#define E_EDGES 625000
#define FIN     128
#define H       128
#define HX      256
#define DE      16
#define G_GRAPHS 64
#define C_OUT   10
#define L_LAYERS 2
#define EPS_GEN 1e-7f
#define EPS_BN  1e-5f

typedef __attribute__((ext_vector_type(16))) _Float16 v16h;
typedef __attribute__((ext_vector_type(8)))  _Float16 v8h;
typedef __attribute__((ext_vector_type(8)))  float    v8f;

// ---------------------------------------------------------------------------
// Order-preserving float->uint encoding so segment-max can use atomicMax(u32).
// Zero (memset) is below every encoded finite float -> valid identity.
// ---------------------------------------------------------------------------
__device__ __forceinline__ unsigned enc_f32(float f) {
  unsigned u = __float_as_uint(f);
  return (u & 0x80000000u) ? ~u : (u | 0x80000000u);
}
__device__ __forceinline__ float dec_f32(unsigned u) {
  return (u & 0x80000000u) ? __uint_as_float(u & 0x7FFFFFFFu)
                           : __uint_as_float(~u);
}

// ---------------------------------------------------------------------------
// WMMA GEMM: D[M,Nc] = (A (+A2)) @ B + bias (+R residual).
//  - Block = 256 threads (8 waves) covering 128 rows x 64 cols of D.
//  - B panel (K x 64) is staged once per block into LDS, f16, TRANSPOSED
//    (column-major, stride K+8 shorts -> conflict-free half-wave reads).
//    Each WMMA B fragment is then two contiguous 16B ds_load_b128 per lane.
//  - Each wave computes 4 adjacent 16x16 tiles, reusing one A fragment.
//  - K multiple of 32; M, Nc multiples of 16; Nc multiple of 64.
// ---------------------------------------------------------------------------
#define BN_COLS 64
#define BS_MAX  (BN_COLS * (HX + 8))   // worst case K=256 -> 16896 shorts

template<bool ADD2, bool RESID>
__global__ void gemm_wmma_f16(const float* __restrict__ A,
                              const float* __restrict__ A2,
                              const float* __restrict__ B,
                              const float* __restrict__ bias,
                              const float* __restrict__ R,
                              float* __restrict__ D,
                              int M, int K, int Nc) {
  __shared__ __align__(16) _Float16 bs[BS_MAX];

  const int colBlocks = Nc / BN_COLS;
  const int rowBlk = blockIdx.x / colBlocks;
  const int colBlk = blockIdx.x % colBlocks;
  const int colBase = colBlk * BN_COLS;
  const int S = K + 8;                       // shorts per LDS column

  // ---- stage B panel (K x 64) into LDS, transposed to [n][k] ----
  for (int idx = threadIdx.x; idx < K * BN_COLS; idx += 256) {
    const int k = idx >> 6;                  // consecutive tid -> consecutive n
    const int n = idx & (BN_COLS - 1);       // (coalesced global reads)
    bs[n * S + k] = (_Float16)B[(size_t)k * Nc + colBase + n];
  }
  __syncthreads();

  const int wave  = threadIdx.x >> 5;
  const int lane  = threadIdx.x & 31;
  const int l15   = lane & 15;
  const int khalf = lane >> 4;
  const int row0  = rowBlk * 128 + wave * 16;
  if (row0 >= M) return;                     // wave-uniform (after barrier)

  // ---- C init: bias (+ residual) for 4 tiles ----
  v8f c[4];
#pragma unroll
  for (int j = 0; j < 4; ++j) {
    const float b0 = bias[colBase + j * 16 + l15];
#pragma unroll
    for (int i = 0; i < 8; ++i) {
      float v = b0;
      if (RESID) {
        const int m = row0 + i + (khalf << 3);
        v += R[(size_t)m * Nc + colBase + j * 16 + l15];
      }
      c[j][i] = v;
    }
  }

  const float* Arow  = A + (size_t)(row0 + l15) * K;
  const float* Arow2 = ADD2 ? (A2 + (size_t)(row0 + l15) * K) : nullptr;

  for (int k0 = 0; k0 < K; k0 += 32) {
    // A 16x32 f16 fragment: VGPR i pair at kb = (i>>2)*16 + khalf*8 + (i&3)*2
    v16h a;
#pragma unroll
    for (int i = 0; i < 8; ++i) {
      const int kb = k0 + ((i >> 2) << 4) + (khalf << 3) + ((i & 3) << 1);
      float2 f = *(const float2*)(Arow + kb);
      if (ADD2) {
        float2 g = *(const float2*)(Arow2 + kb);
        f.x += g.x; f.y += g.y;
      }
      a[2 * i]     = (_Float16)f.x;
      a[2 * i + 1] = (_Float16)f.y;
    }
    // 4 WMMAs share the A fragment; B fragments from LDS (2x ds_load_b128)
#pragma unroll
    for (int j = 0; j < 4; ++j) {
      const _Float16* col = &bs[(j * 16 + l15) * S + k0 + (khalf << 3)];
      const v8h lo = *(const v8h*)(col);
      const v8h hi = *(const v8h*)(col + 16);
      const v16h b = __builtin_shufflevector(lo, hi, 0, 1, 2, 3, 4, 5, 6, 7,
                                             8, 9, 10, 11, 12, 13, 14, 15);
      c[j] = __builtin_amdgcn_wmma_f32_16x16x32_f16(false, a, false, b,
                                                    (short)0, c[j], false, false);
    }
  }

  // ---- store: C/D layout VGPR i -> row m = i + khalf*8, col = l15 ----
#pragma unroll
  for (int j = 0; j < 4; ++j) {
#pragma unroll
    for (int i = 0; i < 8; ++i) {
      const int m = row0 + i + (khalf << 3);
      D[(size_t)m * Nc + colBase + j * 16 + l15] = c[j][i];
    }
  }
}

// ---------------------------------------------------------------------------
// BatchNorm statistics: one thread per channel, grid-stride over rows,
// one atomicAdd per (block, channel). stats[0..Hc)=sum, stats[Hc..2Hc)=sumsq.
// ---------------------------------------------------------------------------
__global__ void bn_stats(const float* __restrict__ in, float* __restrict__ stats,
                         int n, int Hc) {
  const int ch = threadIdx.x;
  float s = 0.f, ss = 0.f;
  for (int r = blockIdx.x; r < n; r += gridDim.x) {
    const float v = in[(size_t)r * Hc + ch];
    s += v; ss += v * v;
  }
  atomicAdd(&stats[ch], s);
  atomicAdd(&stats[Hc + ch], ss);
}

__global__ void bn_apply_relu(const float* __restrict__ in, float* __restrict__ out,
                              const float* __restrict__ stats,
                              const float* __restrict__ gamma,
                              const float* __restrict__ beta,
                              int n, int Hc) {
  const size_t i = (size_t)blockIdx.x * blockDim.x + threadIdx.x;
  if (i >= (size_t)n * Hc) return;
  const int ch = (int)(i % Hc);
  const float mu  = stats[ch] / (float)n;
  const float var = stats[Hc + ch] / (float)n - mu * mu;
  const float v = (in[i] - mu) * rsqrtf(var + EPS_BN) * gamma[ch] + beta[ch];
  out[i] = fmaxf(v, 0.f);
}

// ---------------------------------------------------------------------------
// GENConv edge passes: one wave per edge, 4 channels per lane.
// Recomputes e_proj = edge_attr @ W_e + b_e each pass (W_e is 8 KB, L0-hot)
// instead of materializing a 320 MB E x H tensor.  h / zmax / den / agg are
// all ~25.6 MB -> L2-resident on the 192 MB L2.
// pass 0: segment-max of z=msg*t  (encoded-u32 atomicMax)
// pass 1: den += exp(z - zmax)
// pass 2: agg += msg * exp(z - zmax) / (den + 1e-16)
// ---------------------------------------------------------------------------
__global__ void edge_pass(int pass,
                          const float* __restrict__ h,
                          const float* __restrict__ edge_attr,
                          const int* __restrict__ src,
                          const int* __restrict__ dst,
                          const float* __restrict__ We,
                          const float* __restrict__ be,
                          const float* __restrict__ tptr,
                          unsigned* __restrict__ zmax,
                          float* __restrict__ den,
                          float* __restrict__ agg) {
  const long gid = (long)blockIdx.x * blockDim.x + threadIdx.x;
  const int e = (int)(gid >> 5);
  const int lane = threadIdx.x & 31;
  if (e >= E_EDGES) return;

  const int s = src[e];
  const int d = dst[e];
  const float tval = tptr[0];

  const int cbase = lane * 4;
  float p0 = be[cbase], p1 = be[cbase + 1], p2 = be[cbase + 2], p3 = be[cbase + 3];
  const float* ea = edge_attr + (size_t)e * DE;
#pragma unroll
  for (int k = 0; k < DE; ++k) {
    const float av = ea[k];
    const float* w = We + k * H + cbase;
    p0 += av * w[0]; p1 += av * w[1]; p2 += av * w[2]; p3 += av * w[3];
  }

  const float* hs = h + (size_t)s * H + cbase;     // contiguous 512B per wave
  const float m0 = fmaxf(hs[0] + p0, 0.f) + EPS_GEN;
  const float m1 = fmaxf(hs[1] + p1, 0.f) + EPS_GEN;
  const float m2 = fmaxf(hs[2] + p2, 0.f) + EPS_GEN;
  const float m3 = fmaxf(hs[3] + p3, 0.f) + EPS_GEN;

  const float z0 = m0 * tval, z1 = m1 * tval, z2 = m2 * tval, z3 = m3 * tval;
  const size_t base = (size_t)d * H + cbase;

  if (pass == 0) {
    atomicMax(&zmax[base + 0], enc_f32(z0));
    atomicMax(&zmax[base + 1], enc_f32(z1));
    atomicMax(&zmax[base + 2], enc_f32(z2));
    atomicMax(&zmax[base + 3], enc_f32(z3));
  } else {
    const float ex0 = __expf(z0 - dec_f32(zmax[base + 0]));
    const float ex1 = __expf(z1 - dec_f32(zmax[base + 1]));
    const float ex2 = __expf(z2 - dec_f32(zmax[base + 2]));
    const float ex3 = __expf(z3 - dec_f32(zmax[base + 3]));
    if (pass == 1) {
      atomicAdd(&den[base + 0], ex0);
      atomicAdd(&den[base + 1], ex1);
      atomicAdd(&den[base + 2], ex2);
      atomicAdd(&den[base + 3], ex3);
    } else {
      atomicAdd(&agg[base + 0], m0 * (ex0 / (den[base + 0] + 1e-16f)));
      atomicAdd(&agg[base + 1], m1 * (ex1 / (den[base + 1] + 1e-16f)));
      atomicAdd(&agg[base + 2], m2 * (ex2 / (den[base + 2] + 1e-16f)));
      atomicAdd(&agg[base + 3], m3 * (ex3 / (den[base + 3] + 1e-16f)));
    }
  }
}

// ---------------------------------------------------------------------------
// Global mean pool accumulation: one wave per node, 4 channels per lane.
// ---------------------------------------------------------------------------
__global__ void pool_kernel(const float* __restrict__ x, const int* __restrict__ batch,
                            float* __restrict__ pooled, float* __restrict__ counts) {
  const long gid = (long)blockIdx.x * blockDim.x + threadIdx.x;
  const int n = (int)(gid >> 5);
  const int lane = threadIdx.x & 31;
  if (n >= N_NODES) return;
  const int g = batch[n];
  const float* xr = x + (size_t)n * H + lane * 4;
  atomicAdd(&pooled[(size_t)g * H + lane * 4 + 0], xr[0]);
  atomicAdd(&pooled[(size_t)g * H + lane * 4 + 1], xr[1]);
  atomicAdd(&pooled[(size_t)g * H + lane * 4 + 2], xr[2]);
  atomicAdd(&pooled[(size_t)g * H + lane * 4 + 3], xr[3]);
  if (lane == 0) atomicAdd(&counts[g], 1.0f);
}

// Final head: relu(pooled/count) @ W_out + b_out  ([64,128] @ [128,10] -- tiny)
__global__ void out_head(const float* __restrict__ pooled, const float* __restrict__ counts,
                         const float* __restrict__ W_out, const float* __restrict__ b_out,
                         float* __restrict__ out) {
  const int tid = threadIdx.x;
  if (tid >= G_GRAPHS * C_OUT) return;
  const int g = tid / C_OUT;
  const int c = tid % C_OUT;
  const float cnt = fmaxf(counts[g], 1.0f);
  float acc = b_out[c];
  for (int ch = 0; ch < H; ++ch) {
    const float p = fmaxf(pooled[(size_t)g * H + ch] / cnt, 0.f);
    acc += p * W_out[ch * C_OUT + c];
  }
  out[g * C_OUT + c] = acc;
}

// ---------------------------------------------------------------------------
extern "C" void kernel_launch(void* const* d_in, const int* in_sizes, int n_in,
                              void* d_out, int out_size, void* d_ws, size_t ws_size,
                              hipStream_t stream) {
  const float* x_in  = (const float*)d_in[0];
  const int*   eidx  = (const int*)  d_in[1];
  const float* eattr = (const float*)d_in[2];
  const int*   batch = (const int*)  d_in[3];
  const float* W_in  = (const float*)d_in[4];
  const float* b_in  = (const float*)d_in[5];
  const float* gamma = (const float*)d_in[6];
  const float* beta  = (const float*)d_in[7];
  const float* W_e   = (const float*)d_in[8];
  const float* b_e   = (const float*)d_in[9];
  const float* t     = (const float*)d_in[10];
  const float* W1    = (const float*)d_in[11];
  const float* b1    = (const float*)d_in[12];
  const float* g1    = (const float*)d_in[13];
  const float* bt1   = (const float*)d_in[14];
  const float* W2    = (const float*)d_in[15];
  const float* b2    = (const float*)d_in[16];
  const float* W_out = (const float*)d_in[17];
  const float* b_out = (const float*)d_in[18];

  const int* src = eidx;
  const int* dst = eidx + E_EDGES;

  // Workspace layout (~180 MB)
  float*    ws     = (float*)d_ws;
  float*    x_f    = ws;                                    // N*H
  float*    h_f    = x_f  + (size_t)N_NODES * H;            // N*H
  unsigned* zmax_u = (unsigned*)(h_f + (size_t)N_NODES * H);// N*H
  float*    den_f  = (float*)zmax_u + (size_t)N_NODES * H;  // N*H
  float*    agg_f  = den_f + (size_t)N_NODES * H;           // N*H
  float*    u_f    = agg_f + (size_t)N_NODES * H;           // N*HX
  float*    stats  = u_f   + (size_t)N_NODES * HX;          // 1024
  float*    pooled = stats + 1024;                          // G*H
  float*    counts = pooled + (size_t)G_GRAPHS * H;         // G

  const size_t nhBytes = (size_t)N_NODES * H * sizeof(float);
  const int edgeBlocks = (E_EDGES * 32 + 255) / 256;
  const int rowBlocks  = (N_NODES + 127) / 128;             // 391

  // fc_in: x = x_in @ W_in + b_in
  gemm_wmma_f16<false, false><<<rowBlocks * (H / BN_COLS), 256, 0, stream>>>(
      x_in, nullptr, W_in, b_in, nullptr, x_f, N_NODES, FIN, H);

  for (int l = 0; l < L_LAYERS; ++l) {
    // h = relu(batchnorm(x))
    hipMemsetAsync(stats, 0, 1024 * sizeof(float), stream);
    bn_stats<<<512, H, 0, stream>>>(x_f, stats, N_NODES, H);
    bn_apply_relu<<<(N_NODES * H + 255) / 256, 256, 0, stream>>>(
        x_f, h_f, stats, gamma + l * H, beta + l * H, N_NODES, H);

    // GENConv softmax aggregation (3 edge passes)
    hipMemsetAsync(zmax_u, 0, nhBytes, stream);
    edge_pass<<<edgeBlocks, 256, 0, stream>>>(0, h_f, eattr, src, dst,
        W_e + (size_t)l * DE * H, b_e + l * H, t + l, zmax_u, den_f, agg_f);
    hipMemsetAsync(den_f, 0, nhBytes, stream);
    hipMemsetAsync(agg_f, 0, nhBytes, stream);
    edge_pass<<<edgeBlocks, 256, 0, stream>>>(1, h_f, eattr, src, dst,
        W_e + (size_t)l * DE * H, b_e + l * H, t + l, zmax_u, den_f, agg_f);
    edge_pass<<<edgeBlocks, 256, 0, stream>>>(2, h_f, eattr, src, dst,
        W_e + (size_t)l * DE * H, b_e + l * H, t + l, zmax_u, den_f, agg_f);

    // u = (agg + h) @ W1 + b1   (A2 fusion does the agg+h add)
    gemm_wmma_f16<true, false><<<rowBlocks * (HX / BN_COLS), 256, 0, stream>>>(
        agg_f, h_f, W1 + (size_t)l * H * HX, b1 + l * HX, nullptr,
        u_f, N_NODES, H, HX);

    // u = relu(batchnorm(u))
    hipMemsetAsync(stats, 0, 1024 * sizeof(float), stream);
    bn_stats<<<512, HX, 0, stream>>>(u_f, stats, N_NODES, HX);
    bn_apply_relu<<<(N_NODES * HX + 255) / 256, 256, 0, stream>>>(
        u_f, u_f, stats, g1 + l * HX, bt1 + l * HX, N_NODES, HX);

    // x = x + (u @ W2 + b2)    (residual fused as WMMA C initializer)
    gemm_wmma_f16<false, true><<<rowBlocks * (H / BN_COLS), 256, 0, stream>>>(
        u_f, nullptr, W2 + (size_t)l * HX * H, b2 + l * H, x_f,
        x_f, N_NODES, HX, H);
  }

  // global mean pool + head
  hipMemsetAsync(pooled, 0, (size_t)(G_GRAPHS * H + G_GRAPHS) * sizeof(float), stream);
  pool_kernel<<<(N_NODES * 32 + 255) / 256, 256, 0, stream>>>(x_f, batch, pooled, counts);
  out_head<<<1, 640, 0, stream>>>(pooled, counts, W_out, b_out, (float*)d_out);
}
